// OptimizedECN_88605175316809
// MI455X (gfx1250) — matrix operand order repaired and checked by reading, not compiled, not run on hardware
//
#include <hip/hip_runtime.h>
#include <hip/hip_bf16.h>

typedef __attribute__((ext_vector_type(16))) __bf16 v16bf;
typedef __attribute__((ext_vector_type(8)))  __bf16 v8bf;
typedef __attribute__((ext_vector_type(8)))  float  v8f;
typedef __attribute__((ext_vector_type(4)))  int    v4i;

#define AS1 __attribute__((address_space(1)))
#define AS3 __attribute__((address_space(3)))

#if defined(__has_builtin)
#if __has_builtin(__builtin_amdgcn_global_load_async_to_lds_b128)
#define HAVE_ASYNC_LDS 1
#endif
#endif

__device__ __forceinline__ unsigned short f32_to_bf16(float f) {
  unsigned int u = __float_as_uint(f);
  u += 0x7FFFu + ((u >> 16) & 1u);          // round-to-nearest-even
  return (unsigned short)(u >> 16);
}

// ---------------------------------------------------------------- convert
__global__ void cvt_f32_bf16_kernel(const float* __restrict__ in,
                                    unsigned short* __restrict__ out, int n) {
  int i = blockIdx.x * blockDim.x + threadIdx.x;
  int stride = gridDim.x * blockDim.x;
  for (; i < n; i += stride) out[i] = f32_to_bf16(in[i]);
}

// ---------------------------------------------------------------- gating
// scores = sigmoid(x @ Wg) * route_scale ; top-2 ; normalized combine weights
__global__ void gate_kernel(const float* __restrict__ x, const float* __restrict__ Wg,
                            const float* __restrict__ route_scale,
                            float* __restrict__ combine) {
  const int t = blockIdx.x;          // token
  const int tid = threadIdx.x;       // 128 threads
  float p[8];
#pragma unroll
  for (int e = 0; e < 8; ++e) p[e] = 0.f;
  for (int d = tid; d < 1024; d += 128) {
    const float xv = x[(size_t)t * 1024 + d];
#pragma unroll
    for (int e = 0; e < 8; ++e) p[e] += xv * Wg[d * 8 + e];
  }
  __shared__ float red[128][8];
#pragma unroll
  for (int e = 0; e < 8; ++e) red[tid][e] = p[e];
  __syncthreads();
  for (int s = 64; s > 0; s >>= 1) {
    if (tid < s) {
#pragma unroll
      for (int e = 0; e < 8; ++e) red[tid][e] += red[tid + s][e];
    }
    __syncthreads();
  }
  if (tid == 0) {
    const float rs = route_scale[0];
    float sc[8];
#pragma unroll
    for (int e = 0; e < 8; ++e) sc[e] = rs / (1.f + __expf(-red[0][e]));
    int i0 = 0;
#pragma unroll
    for (int e = 1; e < 8; ++e) if (sc[e] > sc[i0]) i0 = e;
    int i1 = (i0 == 0) ? 1 : 0;
#pragma unroll
    for (int e = 0; e < 8; ++e) if (e != i0 && sc[e] > sc[i1]) i1 = e;
    const float inv = 1.f / (sc[i0] + sc[i1]);
#pragma unroll
    for (int e = 0; e < 8; ++e) combine[(size_t)t * 8 + e] = 0.f;
    combine[(size_t)t * 8 + i0] = sc[i0] * inv;
    combine[(size_t)t * 8 + i1] = sc[i1] * inv;
  }
}

// ---------------------------------------------------------------- WMMA GEMM
// C[M,N] = A[M,K](bf16) @ B[K,N](bf16)  (+bias, gelu, row-scale, accumulate)
// flags: 1 = exact GELU, 2 = accumulate into Cf
// Block: 256 threads = 8 waves, 64x64 output tile; each wave owns a 16x32 strip
// (two 16x16 f32 accumulators sharing one A fragment per 32-wide K slice).
#define KSTEP 128

__global__ __launch_bounds__(256)
void wmma_gemm_bf16(const __bf16* __restrict__ A, const __bf16* __restrict__ B,
                    const float* __restrict__ bias,
                    const float* __restrict__ rowScale, int rsStride,
                    float* Cf, unsigned short* Cbf,
                    int M, int N, int K, int flags) {
  __shared__ __attribute__((aligned(16))) unsigned short ldsA[64 * KSTEP];   // [row][k]
  __shared__ __attribute__((aligned(16))) unsigned short ldsB[64 * KSTEP];   // [col][k] (transposed)
  const int tid  = threadIdx.x;
  const int lane = tid & 31;
  const int w    = tid >> 5;                 // 8 waves -> 4x2 grid of 16x32 strips
  const int mrow = (w >> 1) * 16;
  const int ncol = (w & 1) * 32;
  const int row0 = blockIdx.y * 64;
  const int col0 = blockIdx.x * 64;

  v8f c0 = {}, c1 = {};
  const int m     = lane & 15;               // A row / B col within tile
  const int kb    = (lane >> 4) * 8;         // unused; the A fragment is indexed by khalf
  const int khalf = (lane >> 4) * 16;        // B-fragment K half

  const int ar  = tid >> 2;                  // A stage: row 0..63
  const int ac  = (tid & 3) * 32;            //          k-chunk {0,32,64,96}
  const int bkr = tid >> 1;                  // B stage: k-row 0..127
  const int bcn = (tid & 1) * 32;            //          col-chunk {0,32}

  for (int kt = 0; kt < K; kt += KSTEP) {
    __syncthreads();                         // previous tile fully consumed
    // ---- stage A tile: 64 rows x 128 K (64B/thread)
    {
      const __bf16* gsrc = A + (size_t)(row0 + ar) * K + kt + ac;
      unsigned short* ldst = ldsA + ar * KSTEP + ac;
#ifdef HAVE_ASYNC_LDS
#pragma unroll
      for (int j = 0; j < 4; ++j)
        __builtin_amdgcn_global_load_async_to_lds_b128(
            (AS1 v4i*)(gsrc + j * 8), (AS3 v4i*)(ldst + j * 8), 0, 0);
#else
      const uint4* src = (const uint4*)gsrc;
      uint4* dst = (uint4*)ldst;
      dst[0] = src[0]; dst[1] = src[1]; dst[2] = src[2]; dst[3] = src[3];
#endif
    }
    // ---- stage B tile transposed: 128 K x 64 N -> ldsB[col][k]
    {
      uint4 bvals[4];
      const uint4* src = (const uint4*)(B + (size_t)(kt + bkr) * N + col0 + bcn);
      bvals[0] = src[0]; bvals[1] = src[1]; bvals[2] = src[2]; bvals[3] = src[3];
      const unsigned short* bs = (const unsigned short*)bvals;
#pragma unroll
      for (int j = 0; j < 32; ++j) ldsB[(bcn + j) * KSTEP + bkr] = bs[j];
    }
    if (kt + KSTEP < K) {                    // gfx1250 global_prefetch of next tiles
      __builtin_prefetch(A + (size_t)(row0 + ar) * K + kt + KSTEP + ac, 0, 1);
      __builtin_prefetch(B + (size_t)(kt + KSTEP + bkr) * N + col0 + bcn, 0, 1);
    }
#ifdef HAVE_ASYNC_LDS
#if __has_builtin(__builtin_amdgcn_s_wait_asynccnt)
    __builtin_amdgcn_s_wait_asynccnt(0);
#else
    asm volatile("s_wait_asynccnt 0x0" ::: "memory");
#endif
#endif
    __syncthreads();
    // ---- 4 K-slices x 2 N-tiles = 8 WMMA, A fragment reused across the pair
#pragma unroll
    for (int kk = 0; kk < KSTEP; kk += 32) {
      // A fragment (16x32 bf16): lane<16 -> K {0..15}, lane>=16 -> K {16..31}
      const v8bf a0 = *(const v8bf*)((const __bf16*)ldsA + (mrow + m) * KSTEP + kk + khalf);
      const v8bf a1 = *(const v8bf*)((const __bf16*)ldsA + (mrow + m) * KSTEP + kk + khalf + 8);
      v16bf av;
#pragma unroll
      for (int e = 0; e < 8; ++e) { av[e] = a0[e]; av[8 + e] = a1[e]; }
      // B fragments (32x16 bf16): lane n holds col n; lanes 0-15 K=0..15, 16-31 K=16..31
      const v16bf b0 = *(const v16bf*)((const __bf16*)ldsB + (ncol + m) * KSTEP + kk + khalf);
      const v16bf b1 = *(const v16bf*)((const __bf16*)ldsB + (ncol + 16 + m) * KSTEP + kk + khalf);
      c0 = __builtin_amdgcn_wmma_f32_16x16x32_bf16(false, av, false, b0, (short)0, c0, false, false);
      c1 = __builtin_amdgcn_wmma_f32_16x16x32_bf16(false, av, false, b1, (short)0, c1, false, false);
    }
  }
  // ---- epilogue: C/D layout — VGPR i: lanes 0-15 M=i, lanes 16-31 M=8+i; N = lane&15
  const int gcol0 = col0 + ncol + (lane & 15);
  const int gcol1 = gcol0 + 16;
  const int rbase = row0 + mrow + ((lane >> 4) * 8);
  const float bc0 = bias ? bias[gcol0] : 0.f;
  const float bc1 = bias ? bias[gcol1] : 0.f;
#pragma unroll
  for (int i = 0; i < 8; ++i) {
    const int grow = rbase + i;
    const float rsc = rowScale ? rowScale[(size_t)grow * rsStride] : 1.f;
    float v0 = c0[i] + bc0;
    float v1 = c1[i] + bc1;
    if (flags & 1) {
      v0 = 0.5f * v0 * (1.f + erff(v0 * 0.70710678118654752f));
      v1 = 0.5f * v1 * (1.f + erff(v1 * 0.70710678118654752f));
    }
    v0 *= rsc; v1 *= rsc;
    const size_t idx0 = (size_t)grow * N + gcol0;
    const size_t idx1 = (size_t)grow * N + gcol1;
    if (Cf) {
      if (flags & 2) { Cf[idx0] += v0; Cf[idx1] += v1; }
      else           { Cf[idx0]  = v0; Cf[idx1]  = v1; }
    }
    if (Cbf) { Cbf[idx0] = f32_to_bf16(v0); Cbf[idx1] = f32_to_bf16(v1); }
  }
}

// ---------------------------------------------------------------- attention
// one block of 32 threads per batch; thread = (head, q-pos); S=8, H=4, Dh=32
__global__ void attn_kernel(const float* __restrict__ q, const float* __restrict__ k,
                            const float* __restrict__ v, unsigned short* __restrict__ obf) {
  const int b   = blockIdx.x;
  const int tid = threadIdx.x;
  const int h   = tid >> 3;
  const int qp  = tid & 7;
  const size_t qoff = ((size_t)b * 8 + qp) * 128 + h * 32;
  float qv[32];
#pragma unroll
  for (int d = 0; d < 32; ++d) qv[d] = q[qoff + d];
  // rope on first 16 dims of q only (matches reference)
#pragma unroll
  for (int i = 0; i < 8; ++i) {
    const float freq = __powf(10000.f, -(float)i / 8.f);
    const float ang  = (float)qp * freq;
    const float cs = __cosf(ang), sn = __sinf(ang);
    const float x1 = qv[i], x2 = qv[8 + i];
    qv[i]     = x1 * cs - x2 * sn;
    qv[8 + i] = x1 * sn + x2 * cs;
  }
  const float scale = 0.1767766952966369f;   // 1/sqrt(32)
  float logits[8];
#pragma unroll
  for (int kp = 0; kp < 8; ++kp) {
    const size_t koff = ((size_t)b * 8 + kp) * 128 + h * 32;
    float acc = 0.f;
#pragma unroll
    for (int d = 0; d < 32; ++d) acc += qv[d] * k[koff + d];
    logits[kp] = acc * scale;
  }
  float mx = logits[0];
#pragma unroll
  for (int kp = 1; kp < 8; ++kp) mx = fmaxf(mx, logits[kp]);
  float sum = 0.f;
#pragma unroll
  for (int kp = 0; kp < 8; ++kp) { logits[kp] = __expf(logits[kp] - mx); sum += logits[kp]; }
  const float inv = 1.f / sum;
  float o[32];
#pragma unroll
  for (int d = 0; d < 32; ++d) o[d] = 0.f;
#pragma unroll
  for (int kp = 0; kp < 8; ++kp) {
    const size_t voff = ((size_t)b * 8 + kp) * 128 + h * 32;
    const float wgt = logits[kp] * inv;
#pragma unroll
    for (int d = 0; d < 32; ++d) o[d] += wgt * v[voff + d];
  }
#pragma unroll
  for (int d = 0; d < 32; ++d) obf[qoff + d] = f32_to_bf16(o[d]);
}

// ---------------------------------------------------------------- value head
__global__ void value_kernel(const float* __restrict__ comb, const float* __restrict__ Wv,
                             const float* __restrict__ bv, float* __restrict__ out) {
  const int t = blockIdx.x;
  const int tid = threadIdx.x;   // 256
  float p = 0.f;
  for (int d = tid; d < 1024; d += 256) p += comb[(size_t)t * 1024 + d] * Wv[d];
  __shared__ float red[256];
  red[tid] = p;
  __syncthreads();
  for (int s = 128; s > 0; s >>= 1) {
    if (tid < s) red[tid] += red[tid + s];
    __syncthreads();
  }
  if (tid == 0) out[t] = red[0] + bv[0];
}

// ================================================================ launch
extern "C" void kernel_launch(void* const* d_in, const int* in_sizes, int n_in,
                              void* d_out, int out_size, void* d_ws, size_t ws_size,
                              hipStream_t stream) {
  (void)in_sizes; (void)n_in; (void)out_size; (void)ws_size;
  const int NT = 2048 * 8;   // tokens
  const int D = 1024, F = 1024, E = 8, Lr = 256, HD = 128, O = 1024;

  const float* x   = (const float*)d_in[0];
  const float* Wg  = (const float*)d_in[1];
  const float* W1  = (const float*)d_in[2];
  const float* b1  = (const float*)d_in[3];
  const float* W2  = (const float*)d_in[4];
  const float* b2  = (const float*)d_in[5];
  const float* Ws1 = (const float*)d_in[6];
  const float* bs1 = (const float*)d_in[7];
  const float* Ws2 = (const float*)d_in[8];
  const float* bs2 = (const float*)d_in[9];
  const float* route_scale = (const float*)d_in[10];
  const float* Wq  = (const float*)d_in[11];
  const float* Wkv = (const float*)d_in[12];
  const float* Wku = (const float*)d_in[13];
  const float* Wvu = (const float*)d_in[14];
  const float* Wao = (const float*)d_in[15];
  const float* Wo  = (const float*)d_in[16];
  const float* bo  = (const float*)d_in[17];
  const float* Wv  = (const float*)d_in[18];
  const float* bv  = (const float*)d_in[19];

  char* ws = (char*)d_ws;
  size_t off = 0;
  auto alloc = [&](size_t bytes) -> void* {
    void* p = ws + off;
    off += (bytes + 255) & ~(size_t)255;
    return p;
  };
  unsigned short* xf_bf  = (unsigned short*)alloc((size_t)NT * D * 2);
  unsigned short* W1_bf  = (unsigned short*)alloc((size_t)E * D * F * 2);
  unsigned short* W2_bf  = (unsigned short*)alloc((size_t)E * F * D * 2);
  unsigned short* Ws1_bf = (unsigned short*)alloc((size_t)D * F * 2);
  unsigned short* Ws2_bf = (unsigned short*)alloc((size_t)F * D * 2);
  unsigned short* Wq_bf  = (unsigned short*)alloc((size_t)D * HD * 2);
  unsigned short* Wkv_bf = (unsigned short*)alloc((size_t)D * Lr * 2);
  unsigned short* Wku_bf = (unsigned short*)alloc((size_t)Lr * HD * 2);
  unsigned short* Wvu_bf = (unsigned short*)alloc((size_t)Lr * HD * 2);
  unsigned short* Wao_bf = (unsigned short*)alloc((size_t)HD * D * 2);
  unsigned short* Wo_bf  = (unsigned short*)alloc((size_t)D * O * 2);
  float*          combine= (float*)alloc((size_t)NT * E * 4);
  unsigned short* H_bf   = (unsigned short*)alloc((size_t)NT * F * 2);
  float*          comb_f = (float*)alloc((size_t)NT * D * 4);
  unsigned short* xa_bf  = (unsigned short*)alloc((size_t)NT * D * 2);
  float*          qbuf   = (float*)alloc((size_t)NT * HD * 4);
  unsigned short* lat_bf = (unsigned short*)alloc((size_t)NT * Lr * 2);
  float*          kbuf   = (float*)alloc((size_t)NT * HD * 4);
  float*          vbuf   = (float*)alloc((size_t)NT * HD * 4);
  unsigned short* o_bf   = (unsigned short*)alloc((size_t)NT * HD * 2);

  auto cvt = [&](const float* src, unsigned short* dst, size_t n) {
    int blocks = (int)((n + 255) / 256);
    if (blocks > 8192) blocks = 8192;
    cvt_f32_bf16_kernel<<<dim3(blocks), dim3(256), 0, stream>>>(src, dst, (int)n);
  };
  auto gemm = [&](const unsigned short* A, const unsigned short* B, const float* bias,
                  const float* rowScale, int rsStride, float* Cf, unsigned short* Cbf,
                  int M, int N, int K, int flags) {
    dim3 grid(N / 64, M / 64);
    wmma_gemm_bf16<<<grid, dim3(256), 0, stream>>>(
        (const __bf16*)A, (const __bf16*)B, bias, rowScale, rsStride, Cf, Cbf, M, N, K, flags);
  };

  // ---- precision conversions (weights + input)
  cvt(x,   xf_bf,  (size_t)NT * D);
  cvt(W1,  W1_bf,  (size_t)E * D * F);
  cvt(W2,  W2_bf,  (size_t)E * F * D);
  cvt(Ws1, Ws1_bf, (size_t)D * F);
  cvt(Ws2, Ws2_bf, (size_t)F * D);
  cvt(Wq,  Wq_bf,  (size_t)D * HD);
  cvt(Wkv, Wkv_bf, (size_t)D * Lr);
  cvt(Wku, Wku_bf, (size_t)Lr * HD);
  cvt(Wvu, Wvu_bf, (size_t)Lr * HD);
  cvt(Wao, Wao_bf, (size_t)HD * D);
  cvt(Wo,  Wo_bf,  (size_t)D * O);

  // ---- gating (top-2 combine weights)
  gate_kernel<<<dim3(NT), dim3(128), 0, stream>>>(x, Wg, route_scale, combine);

  // ---- shared FFN: comb_f = gelu(x@Ws1+bs1)@Ws2 + bs2
  gemm(xf_bf, Ws1_bf, bs1, nullptr, 0, nullptr, H_bf, NT, F, D, 1);
  gemm(H_bf,  Ws2_bf, bs2, nullptr, 0, comb_f, nullptr, NT, D, F, 0);

  // ---- experts: comb_f += combine[:,e] * (gelu(x@W1[e]+b1[e])@W2[e] + b2[e])
  for (int e = 0; e < E; ++e) {
    gemm(xf_bf, W1_bf + (size_t)e * D * F, b1 + (size_t)e * F,
         nullptr, 0, nullptr, H_bf, NT, F, D, 1);
    gemm(H_bf, W2_bf + (size_t)e * F * D, b2 + (size_t)e * D,
         combine + e, E, comb_f, nullptr, NT, D, F, 2);
  }

  // ---- MLA on expert output
  cvt(comb_f, xa_bf, (size_t)NT * D);
  gemm(xa_bf, Wq_bf,  nullptr, nullptr, 0, qbuf, nullptr,  NT, HD, D, 0);
  gemm(xa_bf, Wkv_bf, nullptr, nullptr, 0, nullptr, lat_bf, NT, Lr, D, 0);
  gemm(lat_bf, Wku_bf, nullptr, nullptr, 0, kbuf, nullptr,  NT, HD, Lr, 0);
  gemm(lat_bf, Wvu_bf, nullptr, nullptr, 0, vbuf, nullptr,  NT, HD, Lr, 0);
  attn_kernel<<<dim3(2048), dim3(32), 0, stream>>>(qbuf, kbuf, vbuf, o_bf);
  gemm(o_bf, Wao_bf, nullptr, nullptr, 0, comb_f, nullptr, NT, D, HD, 2);  // combined = expert + attn

  // ---- heads
  cvt(comb_f, xa_bf, (size_t)NT * D);
  gemm(xa_bf, Wo_bf, bo, nullptr, 0, (float*)d_out, nullptr, NT, O, D, 0);
  value_kernel<<<dim3(NT), dim3(256), 0, stream>>>(comb_f, Wv, bv, (float*)d_out + (size_t)NT * D);
}
